// SchNetInteraction_old_77017353552163
// MI455X (gfx1250) — matrix operand order repaired and checked by reading, not compiled, not run on hardware
//
#include <hip/hip_runtime.h>
#include <hip/hip_bf16.h>
#include <math.h>

#define N_NODES 50000
#define N_EDGES 600000
#define HD 128
#define GD 50
#define GP 64      // G padded to multiple of 32 for WMMA K
#define FD 128
#define CUTOFF 10.0f

typedef __attribute__((ext_vector_type(16))) __bf16 v16bf;
typedef __attribute__((ext_vector_type(8)))  float  v8f;

__device__ inline v8f wmma_bf16(v16bf a, v16bf b, v8f c) {
  // D = A(16x32 bf16) x B(32x16 bf16) + C(16x16 f32)
  return __builtin_amdgcn_wmma_f32_16x16x32_bf16(false, a, false, b, (short)0, c, false, false);
}

// A-matrix fragment: 16x32 bf16, lane l<16 -> row M=l, K = [0..7,16..23]; lanes 16..31 -> K+8.
__device__ inline v16bf frag_a(const __bf16* base, int ldk, int lane, int kstep) {
  int r  = lane & 15;
  int kb = kstep * 32 + ((lane >> 4) << 3);
  const __bf16* p = base + r * ldk + kb;
  v16bf f;
#pragma unroll
  for (int j = 0; j < 8; ++j) { f[j] = p[j]; f[j + 8] = p[j + 16]; }
  return f;
}

// B-matrix fragment: 32x16 bf16, lane l<16 -> col N=l, K = [0..7,16..23]; lanes 16..31 -> K+8.
// base points at transposed weights wT[n][k] (row-major, row length ldk).
__device__ inline v16bf frag_b(const __bf16* base, int ldk, int lane, int kstep) {
  int n  = lane & 15;
  int kb = kstep * 32 + ((lane >> 4) << 3);
  const __bf16* p = base + n * ldk + kb;
  v16bf f;
#pragma unroll
  for (int j = 0; j < 8; ++j) { f[j] = p[j]; f[j + 8] = p[j + 16]; }
  return f;
}

__device__ inline void wave_ds_fence() {
  // same-wave LDS stage -> fragment-read ordering (also WAR before async reuse)
  asm volatile("s_wait_dscnt 0" ::: "memory");
}

__device__ inline void wait_async() {
  asm volatile("s_wait_asynccnt 0" ::: "memory");
}

// Per-lane gather of 16 bytes global -> LDS through the CDNA5 async path.
// lds_off: wave-relative LDS byte address (low 32 bits of generic pointer).
__device__ inline void async_gather_b128(unsigned lds_off, const void* gptr) {
  asm volatile("global_load_async_to_lds_b128 %0, %1, off"
               :: "v"(lds_off), "v"((unsigned long long)(uintptr_t)gptr)
               : "memory");
}

__device__ inline unsigned lds_addr_of(const void* p) {
  return (unsigned)(uintptr_t)p;
}

__device__ inline void atomicMaxF(float* a, float v) {
  if (v >= 0.f) atomicMax((int*)a, __float_as_int(v));
  else          atomicMin((unsigned int*)a, __float_as_uint(v));
}

__device__ inline float sspf(float v) {  // softplus(x) - log(2)
  float sp = (v > 20.f) ? v : log1pf(expf(v));
  return sp - 0.69314718056f;
}

// Stage fp32 weight w[K][128] (row-major) into LDS as bf16 transposed wT[128][Kpad].
__device__ inline void stage_wT(const float* w, __bf16* dst, int K, int Kpad,
                                int tid, int nth) {
  for (int i = tid; i < 128 * Kpad; i += nth) {
    int n = i / Kpad, k = i - n * Kpad;
    dst[i] = (__bf16)((k < K) ? w[k * 128 + n] : 0.f);
  }
}

// ---------------------------------------------------------------------------
__global__ void k_init(float* segmax, float* segsum, float* agg) {
  int i = blockIdx.x * blockDim.x + threadIdx.x;
  if (i < N_NODES) { segmax[i] = -1e30f; segsum[i] = 0.f; }
  if (i < N_NODES * HD) agg[i] = 0.f;
}

// h (f32) -> hbf (bf16 copy): halves all downstream gather traffic.
__global__ void k_h2bf(const float* __restrict__ h, __bf16* __restrict__ hbf) {
  int i = blockIdx.x * blockDim.x + threadIdx.x;
  if (i < N_NODES * HD) hbf[i] = (__bf16)h[i];
}

// ---------------------------------------------------------------------------
// x = h @ lin1_w   (N x 128 @ 128 x 128), A operands async-copied from hbf
__global__ void __launch_bounds__(128) k_node_x(const __bf16* __restrict__ hbf,
                                                const float* __restrict__ lin1_w,
                                                float* __restrict__ x) {
  __shared__ __align__(16) __bf16 wT[HD * HD];
  __shared__ __align__(16) __bf16 atile[4][16 * HD];
  int tid = threadIdx.x, lane = tid & 31, wid = tid >> 5;
  stage_wT(lin1_w, wT, HD, HD, tid, 128);
  __syncthreads();
  const int ntiles = N_NODES / 16;
  for (int t = blockIdx.x * 4 + wid; t < ntiles; t += gridDim.x * 4) {
    int n0 = t * 16;
    __bf16* at = atile[wid];
    int r = lane & 15, half = lane >> 4;
    wave_ds_fence();  // previous iteration's fragment reads done before overwrite
    {
      const __bf16* src = hbf + (size_t)(n0 + r) * HD + half * 64;
      unsigned dl = lds_addr_of(at + r * HD + half * 64);
#pragma unroll
      for (int k = 0; k < 8; ++k)
        async_gather_b128(dl + k * 16, src + k * 8);
    }
    wait_async();
    for (int cb = 0; cb < 8; ++cb) {
      v8f acc = {};
#pragma unroll
      for (int ks = 0; ks < 4; ++ks)
        acc = wmma_bf16(frag_a(at, HD, lane, ks), frag_b(wT + cb * 16 * HD, HD, lane, ks), acc);
      int n = cb * 16 + (lane & 15), mb = (lane >> 4) * 8;
#pragma unroll
      for (int rr = 0; rr < 8; ++rr)
        x[(size_t)(n0 + mb + rr) * HD + n] = acc[rr];
    }
  }
}

// ---------------------------------------------------------------------------
// Edge pass 1: d, gaussian RBF, hid = relu(h[row]@w_hi + h[col]@w_hj + ea@w_d + b1),
//              score = hid@w2 + b2, atomic seg-max of masked score.
__global__ void __launch_bounds__(128) k_edge1(
    const __bf16* __restrict__ hbf, const float* __restrict__ pos,
    const int* __restrict__ eidx, const unsigned char* __restrict__ mask,
    const float* __restrict__ attn_w1, const float* __restrict__ attn_b1,
    const float* __restrict__ attn_w2, const float* __restrict__ attn_b2,
    float* __restrict__ d_arr, float* __restrict__ score, float* __restrict__ segmax) {
  __shared__ __align__(16) __bf16 wThi[HD * HD];
  __shared__ __align__(16) __bf16 wThj[HD * HD];
  __shared__ __align__(16) __bf16 wTd[HD * GP];
  __shared__ float b1s[HD], w2s[HD];
  __shared__ __align__(16) __bf16 hrow[4][16 * HD];
  __shared__ __align__(16) __bf16 hcol[4][16 * HD];
  __shared__ __align__(16) __bf16 eat[4][16 * GP];
  __shared__ float dls[4][16];
  __shared__ int rls[4][16], cls[4][16];

  int tid = threadIdx.x, lane = tid & 31, wid = tid >> 5;
  stage_wT(attn_w1,               wThi, HD, HD, tid, 128);
  stage_wT(attn_w1 + HD * HD,     wThj, HD, HD, tid, 128);
  stage_wT(attn_w1 + 2 * HD * HD, wTd,  GD, GP, tid, 128);
  for (int i = tid; i < HD; i += 128) { b1s[i] = attn_b1[i]; w2s[i] = attn_w2[i]; }
  __syncthreads();
  const float b2 = attn_b2[0];
  const float delta = CUTOFF / (GD - 1);
  const float coeff = -0.5f / (delta * delta);

  const int ntiles = N_EDGES / 16;
  for (int t = blockIdx.x * 4 + wid; t < ntiles; t += gridDim.x * 4) {
    int e0 = t * 16;
    wave_ds_fence();  // previous iteration fully consumed before restage
    if (lane < 16) {
      int e = e0 + lane;
      int r = eidx[e], c = eidx[N_EDGES + e];
      float dx = pos[r * 3 + 0] - pos[c * 3 + 0];
      float dy = pos[r * 3 + 1] - pos[c * 3 + 1];
      float dz = pos[r * 3 + 2] - pos[c * 3 + 2];
      float d = sqrtf(dx * dx + dy * dy + dz * dz);
      d_arr[e] = d;
      dls[wid][lane] = d; rls[wid][lane] = r; cls[wid][lane] = c;
    }
    wave_ds_fence();
    {
      int rr = lane & 15, half = lane >> 4;
      int sr = rls[wid][rr], sc = cls[wid][rr];
      const __bf16* ps = hbf + (size_t)sr * HD + half * 64;
      const __bf16* pc = hbf + (size_t)sc * HD + half * 64;
      unsigned ldr = lds_addr_of(hrow[wid] + rr * HD + half * 64);
      unsigned ldc = lds_addr_of(hcol[wid] + rr * HD + half * 64);
#pragma unroll
      for (int k = 0; k < 8; ++k) {
        async_gather_b128(ldr + k * 16, ps + k * 8);
        async_gather_b128(ldc + k * 16, pc + k * 8);
      }
      // overlap: Gaussian RBF tile while the async gather is in flight
      float d = dls[wid][rr];
      __bf16* pe = eat[wid] + rr * GP + half * 32;
      for (int g = 0; g < 32; ++g) {
        int gg = half * 32 + g;
        float off = gg * delta;
        float v = (gg < GD) ? expf(coeff * (d - off) * (d - off)) : 0.f;
        pe[g] = (__bf16)v;
      }
    }
    wait_async();
    wave_ds_fence();
    float sp[8] = {0, 0, 0, 0, 0, 0, 0, 0};
    for (int cb = 0; cb < 8; ++cb) {
      float bb = b1s[cb * 16 + (lane & 15)];
      v8f acc;
#pragma unroll
      for (int r2 = 0; r2 < 8; ++r2) acc[r2] = bb;
#pragma unroll
      for (int ks = 0; ks < 4; ++ks) {
        acc = wmma_bf16(frag_a(hrow[wid], HD, lane, ks), frag_b(wThi + cb * 16 * HD, HD, lane, ks), acc);
        acc = wmma_bf16(frag_a(hcol[wid], HD, lane, ks), frag_b(wThj + cb * 16 * HD, HD, lane, ks), acc);
      }
#pragma unroll
      for (int ks = 0; ks < 2; ++ks)
        acc = wmma_bf16(frag_a(eat[wid], GP, lane, ks), frag_b(wTd + cb * 16 * GP, GP, lane, ks), acc);
      float w2v = w2s[cb * 16 + (lane & 15)];
#pragma unroll
      for (int r2 = 0; r2 < 8; ++r2) sp[r2] += fmaxf(acc[r2], 0.f) * w2v;
    }
#pragma unroll
    for (int r2 = 0; r2 < 8; ++r2) {
      float v = sp[r2];
      v += __shfl_xor(v, 1, 32);
      v += __shfl_xor(v, 2, 32);
      v += __shfl_xor(v, 4, 32);
      v += __shfl_xor(v, 8, 32);
      sp[r2] = v;
    }
    if ((lane & 15) == 0) {
      int hi = lane >> 4;
#pragma unroll
      for (int r2 = 0; r2 < 8; ++r2) {
        int m = r2 + hi * 8, e = e0 + m;
        float s = sp[r2] + b2;
        score[e] = s;
        float mv = mask[e] ? s : -1e30f;
        atomicMaxF(&segmax[rls[wid][m]], mv);
      }
    }
  }
}

// ---------------------------------------------------------------------------
__global__ void k_edge2(const int* __restrict__ eidx, const unsigned char* __restrict__ mask,
                        const float* __restrict__ score, const float* __restrict__ segmax,
                        float* __restrict__ segsum) {
  int e = blockIdx.x * blockDim.x + threadIdx.x;
  if (e >= N_EDGES || !mask[e]) return;
  int r = eidx[e];
  float m = segmax[r]; m = (m < -1e29f) ? 0.f : m;
  atomicAdd(&segsum[r], expf(score[e] - m));
}

// ---------------------------------------------------------------------------
// Edge pass 3: decay-scaled RBF -> filter MLP (WMMA) -> msg = x[row]*W -> atomic scatter to agg[col]
__global__ void __launch_bounds__(128) k_edge3(
    const int* __restrict__ eidx, const unsigned char* __restrict__ mask,
    const float* __restrict__ fw1, const float* __restrict__ fb1,
    const float* __restrict__ fw2, const float* __restrict__ fb2,
    const float* __restrict__ d_arr, const float* __restrict__ score,
    const float* __restrict__ segmax, const float* __restrict__ segsum,
    const float* __restrict__ x, float* __restrict__ agg) {
  __shared__ __align__(16) __bf16 w1T[FD * GP];
  __shared__ __align__(16) __bf16 w2T[FD * FD];
  __shared__ float b1s[FD], b2s[FD];
  __shared__ __align__(16) __bf16 eat[4][16 * GP];
  __shared__ __align__(16) __bf16 t1t[4][16 * FD];
  __shared__ float Cls[4][16], decls[4][16], dls[4][16];
  __shared__ int rls[4][16], cls[4][16];

  int tid = threadIdx.x, lane = tid & 31, wid = tid >> 5;
  stage_wT(fw1, w1T, GD, GP, tid, 128);
  stage_wT(fw2, w2T, FD, FD, tid, 128);
  for (int i = tid; i < FD; i += 128) { b1s[i] = fb1[i]; b2s[i] = fb2[i]; }
  __syncthreads();
  const float delta = CUTOFF / (GD - 1);
  const float coeff = -0.5f / (delta * delta);
  const float pic = 3.14159265358979f / CUTOFF;

  const int ntiles = N_EDGES / 16;
  for (int t = blockIdx.x * 4 + wid; t < ntiles; t += gridDim.x * 4) {
    int e0 = t * 16;
    if (lane < 16) {
      int e = e0 + lane;
      int r = eidx[e], c = eidx[N_EDGES + e];
      float d = d_arr[e];
      float m = segmax[r]; m = (m < -1e29f) ? 0.f : m;
      float ss = segsum[r]; float den = (ss > 0.f) ? ss : 1.f;
      decls[wid][lane] = mask[e] ? expf(score[e] - m) / den : 1.f;
      Cls[wid][lane] = 0.5f * (cosf(d * pic) + 1.f);
      dls[wid][lane] = d; rls[wid][lane] = r; cls[wid][lane] = c;
    }
    wave_ds_fence();
    {
      int rr = lane & 15, half = lane >> 4;
      float d = dls[wid][rr], dc = decls[wid][rr];
      __bf16* pe = eat[wid] + rr * GP + half * 32;
      for (int g = 0; g < 32; ++g) {
        int gg = half * 32 + g;
        float off = gg * delta;
        float v = (gg < GD) ? expf(coeff * (d - off) * (d - off)) * dc : 0.f;
        pe[g] = (__bf16)v;
      }
    }
    wave_ds_fence();
    for (int cb = 0; cb < 8; ++cb) {
      float bb = b1s[cb * 16 + (lane & 15)];
      v8f acc;
#pragma unroll
      for (int r2 = 0; r2 < 8; ++r2) acc[r2] = bb;
#pragma unroll
      for (int ks = 0; ks < 2; ++ks)
        acc = wmma_bf16(frag_a(eat[wid], GP, lane, ks), frag_b(w1T + cb * 16 * GP, GP, lane, ks), acc);
      int n = cb * 16 + (lane & 15), mb = (lane >> 4) * 8;
#pragma unroll
      for (int r2 = 0; r2 < 8; ++r2)
        t1t[wid][(mb + r2) * FD + n] = (__bf16)sspf(acc[r2]);
    }
    wave_ds_fence();
    for (int cb = 0; cb < 8; ++cb) {
      float bb = b2s[cb * 16 + (lane & 15)];
      v8f acc;
#pragma unroll
      for (int r2 = 0; r2 < 8; ++r2) acc[r2] = bb;
#pragma unroll
      for (int ks = 0; ks < 4; ++ks)
        acc = wmma_bf16(frag_a(t1t[wid], FD, lane, ks), frag_b(w2T + cb * 16 * FD, FD, lane, ks), acc);
      int n = cb * 16 + (lane & 15), hi = lane >> 4;
#pragma unroll
      for (int r2 = 0; r2 < 8; ++r2) {
        int m = r2 + hi * 8;
        float Wv = acc[r2] * Cls[wid][m];
        float msg = x[(size_t)rls[wid][m] * HD + n] * Wv;
        atomicAdd(&agg[(size_t)cls[wid][m] * HD + n], msg);
      }
    }
  }
}

// ---------------------------------------------------------------------------
// out = ssp(agg @ lin2_w + lin2_b) @ lin_w + lin_b
__global__ void __launch_bounds__(128) k_node_out(
    const float* __restrict__ agg, const float* __restrict__ lin2_w,
    const float* __restrict__ lin2_b, const float* __restrict__ lin_w,
    const float* __restrict__ lin_b, float* __restrict__ out) {
  __shared__ __align__(16) __bf16 w2T[FD * HD];
  __shared__ __align__(16) __bf16 w3T[HD * HD];
  __shared__ float b2s[HD], b3s[HD];
  __shared__ __align__(16) __bf16 at[4][16 * FD];
  __shared__ __align__(16) __bf16 tt[4][16 * HD];
  int tid = threadIdx.x, lane = tid & 31, wid = tid >> 5;
  stage_wT(lin2_w, w2T, FD, FD, tid, 128);
  stage_wT(lin_w,  w3T, HD, HD, tid, 128);
  for (int i = tid; i < HD; i += 128) { b2s[i] = lin2_b[i]; b3s[i] = lin_b[i]; }
  __syncthreads();
  const int ntiles = N_NODES / 16;
  for (int t = blockIdx.x * 4 + wid; t < ntiles; t += gridDim.x * 4) {
    int n0 = t * 16;
    {
      int r = lane & 15, half = lane >> 4;
      const float* src = agg + (size_t)(n0 + r) * HD + half * 64;
      __bf16* dp = at[wid] + r * FD + half * 64;
#pragma unroll 8
      for (int j = 0; j < 64; ++j) dp[j] = (__bf16)src[j];
    }
    wave_ds_fence();
    for (int cb = 0; cb < 8; ++cb) {
      float bb = b2s[cb * 16 + (lane & 15)];
      v8f acc;
#pragma unroll
      for (int r2 = 0; r2 < 8; ++r2) acc[r2] = bb;
#pragma unroll
      for (int ks = 0; ks < 4; ++ks)
        acc = wmma_bf16(frag_a(at[wid], FD, lane, ks), frag_b(w2T + cb * 16 * FD, FD, lane, ks), acc);
      int n = cb * 16 + (lane & 15), mb = (lane >> 4) * 8;
#pragma unroll
      for (int r2 = 0; r2 < 8; ++r2)
        tt[wid][(mb + r2) * HD + n] = (__bf16)sspf(acc[r2]);
    }
    wave_ds_fence();
    for (int cb = 0; cb < 8; ++cb) {
      float bb = b3s[cb * 16 + (lane & 15)];
      v8f acc;
#pragma unroll
      for (int r2 = 0; r2 < 8; ++r2) acc[r2] = bb;
#pragma unroll
      for (int ks = 0; ks < 4; ++ks)
        acc = wmma_bf16(frag_a(tt[wid], HD, lane, ks), frag_b(w3T + cb * 16 * HD, HD, lane, ks), acc);
      int n = cb * 16 + (lane & 15), mb = (lane >> 4) * 8;
#pragma unroll
      for (int r2 = 0; r2 < 8; ++r2)
        out[(size_t)(n0 + mb + r2) * HD + n] = acc[r2];
    }
  }
}

// ---------------------------------------------------------------------------
__global__ void k_copy_pos(const float* __restrict__ pos, float* __restrict__ dst) {
  int i = blockIdx.x * blockDim.x + threadIdx.x;
  if (i < N_NODES * 3) dst[i] = pos[i];
}

// ---------------------------------------------------------------------------
extern "C" void kernel_launch(void* const* d_in, const int* in_sizes, int n_in,
                              void* d_out, int out_size, void* d_ws, size_t ws_size,
                              hipStream_t stream) {
  const float* h      = (const float*)d_in[0];
  const float* pos    = (const float*)d_in[1];
  const int*   eidx   = (const int*)d_in[2];
  const unsigned char* mask = (const unsigned char*)d_in[3];
  const float* attn_w1 = (const float*)d_in[4];
  const float* attn_b1 = (const float*)d_in[5];
  const float* attn_w2 = (const float*)d_in[6];
  const float* attn_b2 = (const float*)d_in[7];
  const float* fw1    = (const float*)d_in[8];
  const float* fb1    = (const float*)d_in[9];
  const float* fw2    = (const float*)d_in[10];
  const float* fb2    = (const float*)d_in[11];
  const float* lin1_w = (const float*)d_in[12];
  const float* lin2_w = (const float*)d_in[13];
  const float* lin2_b = (const float*)d_in[14];
  const float* lin_w  = (const float*)d_in[15];
  const float* lin_b  = (const float*)d_in[16];

  float* wsf    = (float*)d_ws;
  float* d_arr  = wsf;                        // E
  float* score  = d_arr + N_EDGES;            // E
  float* segmax = score + N_EDGES;            // N
  float* segsum = segmax + N_NODES;           // N
  float* x      = segsum + N_NODES;           // N*HD
  float* agg    = x + (size_t)N_NODES * HD;   // N*HD
  __bf16* hbf   = (__bf16*)(agg + (size_t)N_NODES * HD);  // N*HD bf16

  float* outp = (float*)d_out;

  k_init<<<(N_NODES * HD + 255) / 256, 256, 0, stream>>>(segmax, segsum, agg);
  k_h2bf<<<(N_NODES * HD + 255) / 256, 256, 0, stream>>>(h, hbf);
  k_node_x<<<512, 128, 0, stream>>>(hbf, lin1_w, x);
  k_edge1<<<1024, 128, 0, stream>>>(hbf, pos, eidx, mask, attn_w1, attn_b1, attn_w2,
                                    attn_b2, d_arr, score, segmax);
  k_edge2<<<(N_EDGES + 255) / 256, 256, 0, stream>>>(eidx, mask, score, segmax, segsum);
  k_edge3<<<1024, 128, 0, stream>>>(eidx, mask, fw1, fb1, fw2, fb2, d_arr, score,
                                    segmax, segsum, x, agg);
  k_node_out<<<512, 128, 0, stream>>>(agg, lin2_w, lin2_b, lin_w, lin_b, outp);
  k_copy_pos<<<(N_NODES * 3 + 255) / 256, 256, 0, stream>>>(pos, outp + (size_t)N_NODES * HD);
}
